// Seq2SeqEncoder_28535762715219
// MI455X (gfx1250) — compile-verified, matrix operand order checked by reading, not run on hardware
//
#include <hip/hip_runtime.h>
#include <hip/hip_bf16.h>

// ---------------------------------------------------------------------------
// Types for CDNA5 WMMA (wave32): 16x16x32 f16 -> f32
// ---------------------------------------------------------------------------
typedef __attribute__((ext_vector_type(8)))  _Float16 v8h;
typedef __attribute__((ext_vector_type(16))) _Float16 v16h;
typedef __attribute__((ext_vector_type(8)))  float    v8f;

// Load one 16x32 f16 fragment (A or B) for v_wmma_f32_16x16x32_f16.
// Layout (ISA 7.12.2, 16-bit A 16x32): lane<16 holds row=lane, K in {0..7,16..23};
// lane>=16 holds row=lane-16, K in {8..15,24..31}. Two contiguous 16B loads.
// Works for global or LDS pointers (address space inferred by the compiler).
__device__ __forceinline__ v16h load_frag(const _Float16* base, int ld, int lane) {
  const int r  = lane & 15;
  const int hi = lane >> 4;
  const _Float16* p = base + (size_t)r * ld + (hi << 3);
  v8h lo = *(const v8h*)p;
  v8h hh = *(const v8h*)(p + 16);
  v16h out;
#pragma unroll
  for (int i = 0; i < 8; ++i) { out[i] = lo[i]; out[i + 8] = hh[i]; }
  return out;
}

__device__ __forceinline__ v8f wmma_f16(v16h a, v16h b, v8f c) {
  return __builtin_amdgcn_wmma_f32_16x16x32_f16(false, a, false, b, (short)0, c, false, false);
}

__device__ __forceinline__ float sigmoidf(float x) { return 1.0f / (1.0f + __expf(-x)); }

// ---------------------------------------------------------------------------
// Weight convert + transpose: in fp32 [K][N] row-major -> out f16 [N][K]
// ---------------------------------------------------------------------------
__global__ void transpose_f32_to_f16_kernel(const float* __restrict__ in,
                                            _Float16* __restrict__ out, int K, int N) {
  size_t idx = (size_t)blockIdx.x * blockDim.x + threadIdx.x;
  if (idx >= (size_t)K * N) return;
  int k = (int)(idx / N);
  int n = (int)(idx % N);
  out[(size_t)n * K + k] = (_Float16)in[idx];
}

// ---------------------------------------------------------------------------
// Embedding gather: xs[s][b][e] = embed[src[b][s]][e]  (f32 -> f16)
// ---------------------------------------------------------------------------
__global__ void embed_kernel(const int* __restrict__ src, const float* __restrict__ embed,
                             _Float16* __restrict__ xs, int S, int B, int E) {
  size_t idx = (size_t)blockIdx.x * blockDim.x + threadIdx.x;
  size_t total = (size_t)S * B * E;
  if (idx >= total) return;
  int e = (int)(idx % E);
  size_t sb = idx / E;
  int b = (int)(sb % B);
  int s = (int)(sb / B);
  int tok = src[(size_t)b * S + s];
  xs[idx] = (_Float16)embed[(size_t)tok * E + e];
}

// ---------------------------------------------------------------------------
// WMMA GEMM with LDS double-buffering:
//   C[M,N] = A[M,K](f16, ld=lda) x Bt[N,K](f16) + bias
// Workgroup tile 64x128, K-step 32. A tile (64x32) and B tile (128x32) staged
// through padded LDS (row stride 40 halves -> conflict-free fragment reads).
// Global loads for tile t+1 issued before compute on tile t (1 barrier/step).
// All 5 fragments (1xA, 4xB) loaded BEFORE the WMMA chain so the compiler can
// batch the ds_loads and use graduated s_wait_dscnt thresholds.
// mode 0: f16 store at out[m*ldo + n]
// mode 1: f32 store at out[(b*Sdim + s)*ldo + n], m = s*Bdim + b (fused transpose)
// ---------------------------------------------------------------------------
#define LDT 40  // padded LDS row stride in halves (80B -> 16 distinct bank groups)

__global__ __launch_bounds__(256)
void gemm_bias_kernel(const _Float16* __restrict__ A, const _Float16* __restrict__ Bt,
                      const float* __restrict__ bias, void* __restrict__ out,
                      int M, int N, int K, int lda, int ldo, int mode, int Sdim, int Bdim) {
  __shared__ _Float16 sA[2][64 * LDT];
  __shared__ _Float16 sB[2][128 * LDT];

  const int tid  = threadIdx.x;
  const int wave = tid >> 5;
  const int lane = tid & 31;
  const int hr   = lane & 15;
  const int hi   = lane >> 4;
  const int mt   = wave & 3;   // row tile 0..3
  const int cg   = wave >> 2;  // column half 0..1
  const int m0   = blockIdx.y * 64;
  const int n0   = blockIdx.x * 128;

  // staging indices: A = 256 chunks of 8 halves (64 rows x 4), B = 512 chunks (128 rows x 4)
  const int ar = tid >> 2, ak = (tid & 3) << 3;
  const int br0 = tid >> 2, bk0 = (tid & 3) << 3;   // chunk id = tid
  const int br1 = br0 + 64, bk1 = bk0;              // chunk id = tid + 256

  v8f acc[4];
#pragma unroll
  for (int ct = 0; ct < 4; ++ct)
#pragma unroll
    for (int r = 0; r < 8; ++r) acc[ct][r] = 0.0f;

  const int T = K >> 5;
  v8h ra, rb0, rb1;

  // prologue: tile 0 -> regs -> LDS buffer 0
  {
    const int kk = 0;
    ra  = *(const v8h*)(A  + (size_t)(m0 + ar)  * lda + kk + ak);
    rb0 = *(const v8h*)(Bt + (size_t)(n0 + br0) * K   + kk + bk0);
    rb1 = *(const v8h*)(Bt + (size_t)(n0 + br1) * K   + kk + bk1);
    *(v8h*)&sA[0][ar  * LDT + ak ] = ra;
    *(v8h*)&sB[0][br0 * LDT + bk0] = rb0;
    *(v8h*)&sB[0][br1 * LDT + bk1] = rb1;
  }

  for (int t = 0; t < T; ++t) {
    __syncthreads();  // buf[t&1] stores visible; prior reads of buf[(t+1)&1] done
    const int cur = t & 1;

    v8h na, nb0, nb1;
    if (t + 1 < T) {  // issue next-tile global loads; in flight during compute
      const int kk = (t + 1) << 5;
      na  = *(const v8h*)(A  + (size_t)(m0 + ar)  * lda + kk + ak);
      nb0 = *(const v8h*)(Bt + (size_t)(n0 + br0) * K   + kk + bk0);
      nb1 = *(const v8h*)(Bt + (size_t)(n0 + br1) * K   + kk + bk1);
    }

    // load all fragments first, then the WMMA chain
    v16h a  = load_frag(&sA[cur][(mt * 16) * LDT], LDT, lane);
    v16h b0 = load_frag(&sB[cur][(cg * 64 +  0) * LDT], LDT, lane);
    v16h b1 = load_frag(&sB[cur][(cg * 64 + 16) * LDT], LDT, lane);
    v16h b2 = load_frag(&sB[cur][(cg * 64 + 32) * LDT], LDT, lane);
    v16h b3 = load_frag(&sB[cur][(cg * 64 + 48) * LDT], LDT, lane);
    acc[0] = wmma_f16(a, b0, acc[0]);
    acc[1] = wmma_f16(a, b1, acc[1]);
    acc[2] = wmma_f16(a, b2, acc[2]);
    acc[3] = wmma_f16(a, b3, acc[3]);

    if (t + 1 < T) {
      *(v8h*)&sA[cur ^ 1][ar  * LDT + ak ] = na;
      *(v8h*)&sB[cur ^ 1][br0 * LDT + bk0] = nb0;
      *(v8h*)&sB[cur ^ 1][br1 * LDT + bk1] = nb1;
    }
  }

#pragma unroll
  for (int ct = 0; ct < 4; ++ct) {
    const int nn0 = n0 + cg * 64 + ct * 16;
    const float bn = bias ? bias[nn0 + hr] : 0.0f;
#pragma unroll
    for (int r = 0; r < 8; ++r) {
      const int m = r + 8 * hi;               // C layout: M = r + 8*(lane/16), N = lane&15
      const float v = acc[ct][r] + bn;
      if (mode == 0) {
        ((_Float16*)out)[(size_t)(m0 + mt * 16 + m) * ldo + nn0 + hr] = (_Float16)v;
      } else {
        const int row = m0 + mt * 16 + m;
        const int ss = row / Bdim;
        const int bb = row % Bdim;
        ((float*)out)[((size_t)bb * Sdim + ss) * ldo + nn0 + hr] = v;
      }
    }
  }
}

// ---------------------------------------------------------------------------
// Zero h ping-pong state + barrier counters
// ---------------------------------------------------------------------------
__global__ void init_state_kernel(_Float16* __restrict__ hstate, unsigned int* __restrict__ cnt, int n) {
  int i = blockIdx.x * blockDim.x + threadIdx.x;
  if (i < n) hstate[i] = (_Float16)0.0f;
  if (i < 2) cnt[i] = 0u;
}

// ---------------------------------------------------------------------------
// Persistent bidirectional LSTM recurrence.
// grid = (16, 2): blockIdx.y = direction, 16 workgroups x 32 hidden cols each.
// The workgroup's W slice (4 gates x 32 cols x 512 K = 128KB, padded row stride
// 520 halves) is preloaded ONCE into dynamic LDS and reused for all 512 steps;
// per step only h_prev (64KB/dir) travels the global path. Gate math + cell
// state c stay in registers (one 16x16 position per wave, static mapping).
// Direction-wide sync via monotonic atomic barrier (stand-in for cluster
// barriers, which HIP cannot launch portably).
// ---------------------------------------------------------------------------
#define GPD 16          // workgroups per direction
#define WLD 520         // padded LDS row stride for W slice (1040B = 65*16B)

__global__ __launch_bounds__(256)
void lstm_recur_kernel(const _Float16* __restrict__ gx,    // [2][S*B][2048]
                       const _Float16* __restrict__ Wt_f,  // [2048][512] f16 (transposed W)
                       const _Float16* __restrict__ Wt_b,
                       _Float16* __restrict__ hstate,      // [2][2][B][512]
                       _Float16* __restrict__ hcat,        // [S*B][1024]
                       unsigned int* __restrict__ cnt,     // [2]
                       int S, int B) {
  extern __shared__ _Float16 sW[];                         // [128][WLD]

  const int dir  = blockIdx.y;
  const int j0   = blockIdx.x * 32;
  const int wave = threadIdx.x >> 5;
  const int lane = threadIdx.x & 31;
  const int hr   = lane & 15;
  const int hi   = lane >> 4;
  const int mrow = (wave & 3) * 16;       // M offset within B=64
  const int ntoff = (wave >> 2) * 16;     // 0 or 16 within the 32-col slice
  const int ncol = j0 + ntoff;            // hidden-unit offset within H=512

  const _Float16* Wt  = dir ? Wt_b : Wt_f;
  const _Float16* gxd = gx + (size_t)dir * S * B * 2048;
  _Float16* hs_base   = hstate + (size_t)dir * 2 * B * 512;

  // ---- preload W slice into LDS: 128 rows (gate*32 + c) x 512 K ----
  for (int cid = threadIdx.x; cid < 128 * 64; cid += 256) {
    const int row  = cid >> 6;            // 0..127
    const int koff = (cid & 63) << 3;     // 0..504 step 8
    const int gate = row >> 5;
    const int c    = row & 31;
    *(v8h*)&sW[row * WLD + koff] =
        *(const v8h*)(Wt + (size_t)(gate * 512 + j0 + c) * 512 + koff);
  }
  __syncthreads();

  float cst[8];
#pragma unroll
  for (int r = 0; r < 8; ++r) cst[r] = 0.0f;

  unsigned int bstep = 0;

  for (int s = 0; s < S; ++s) {
    const int t = dir ? (S - 1 - s) : s;
    const _Float16* hprev = hs_base + (size_t)(s & 1) * B * 512;
    _Float16* hnext = hs_base + (size_t)((s + 1) & 1) * B * 512;

    // seed accumulators from precomputed gx (lands exactly in C-layout slots)
    v8f acc[4];
#pragma unroll
    for (int gate = 0; gate < 4; ++gate) {
      const _Float16* gp = gxd + ((size_t)t * B + mrow) * 2048 + gate * 512 + ncol + hr;
#pragma unroll
      for (int r = 0; r < 8; ++r) acc[gate][r] = (float)gp[(size_t)(r + 8 * hi) * 2048];
    }

    // h_prev @ W over K = 512 ; A from global, B from LDS-resident W slice.
    // Fragments loaded before the WMMA chain; unroll x2 keeps two iterations'
    // loads in flight under the matrix ops.
#pragma unroll 2
    for (int kk = 0; kk < 512; kk += 32) {
      v16h a  = load_frag(hprev + (size_t)mrow * 512 + kk, 512, lane);
      v16h b0 = load_frag(&sW[(0 * 32 + ntoff) * WLD + kk], WLD, lane);
      v16h b1 = load_frag(&sW[(1 * 32 + ntoff) * WLD + kk], WLD, lane);
      v16h b2 = load_frag(&sW[(2 * 32 + ntoff) * WLD + kk], WLD, lane);
      v16h b3 = load_frag(&sW[(3 * 32 + ntoff) * WLD + kk], WLD, lane);
      acc[0] = wmma_f16(a, b0, acc[0]);
      acc[1] = wmma_f16(a, b1, acc[1]);
      acc[2] = wmma_f16(a, b2, acc[2]);
      acc[3] = wmma_f16(a, b3, acc[3]);
    }

    // gate math (torch cell order f,i,o,c), c kept in registers
#pragma unroll
    for (int r = 0; r < 8; ++r) {
      const float fg = sigmoidf(acc[0][r]);
      const float ig = sigmoidf(acc[1][r]);
      const float og = sigmoidf(acc[2][r]);
      const float cc = tanhf(acc[3][r]);
      const float cn = fg * cst[r] + ig * cc;
      cst[r] = cn;
      const float h = og * tanhf(cn);
      const int m = r + 8 * hi;
      hnext[(size_t)(mrow + m) * 512 + ncol + hr] = (_Float16)h;
      hcat[((size_t)t * B + mrow + m) * 1024 + dir * 512 + ncol + hr] = (_Float16)h;
    }

    // direction-wide barrier (GPD workgroups), monotonic counter
    __threadfence();
    __syncthreads();
    ++bstep;
    if (threadIdx.x == 0) {
      atomicAdd(&cnt[dir], 1u);
      const unsigned int target = (unsigned int)GPD * bstep;
      while (atomicAdd(&cnt[dir], 0u) < target) __builtin_amdgcn_s_sleep(1);
    }
    __syncthreads();
    __threadfence();
  }
}

// Second output: hcat at t = S-1 (layer 1), f16 -> f32
__global__ void last_hcat_kernel(const _Float16* __restrict__ hcat, float* __restrict__ out2,
                                 int S, int B) {
  int idx = blockIdx.x * blockDim.x + threadIdx.x;
  if (idx >= B * 1024) return;
  out2[idx] = (float)hcat[(size_t)(S - 1) * B * 1024 + idx];
}

// ---------------------------------------------------------------------------
// Host side
// ---------------------------------------------------------------------------
extern "C" void kernel_launch(void* const* d_in, const int* in_sizes, int n_in,
                              void* d_out, int out_size, void* d_ws, size_t ws_size,
                              hipStream_t stream) {
  (void)in_sizes; (void)n_in; (void)out_size; (void)ws_size;
  constexpr int S = 512, B = 64, E = 512;
  constexpr int M = S * B;               // 32768 rows
  const size_t SB = (size_t)S * B;

  // ---- inputs (setup_inputs order) ----
  const int*   src   = (const int*)  d_in[0];
  const float* embed = (const float*)d_in[1];
  const float* W0f = (const float*)d_in[2],  *U0f = (const float*)d_in[3],  *b0f = (const float*)d_in[4];
  const float* W0b = (const float*)d_in[5],  *U0b = (const float*)d_in[6],  *b0b = (const float*)d_in[7];
  const float* Wy0 = (const float*)d_in[8],  *by0 = (const float*)d_in[9];
  const float* W1f = (const float*)d_in[10], *U1f = (const float*)d_in[11], *b1f = (const float*)d_in[12];
  const float* W1b = (const float*)d_in[13], *U1b = (const float*)d_in[14], *b1b = (const float*)d_in[15];
  const float* Wy1 = (const float*)d_in[16], *by1 = (const float*)d_in[17];

  // ---- workspace carve-out (256B aligned) ----
  size_t off = 0;
  auto carve = [&](size_t bytes) -> char* {
    char* p = (char*)d_ws + off;
    off += (bytes + 255) & ~(size_t)255;
    return p;
  };
  _Float16* xs     = (_Float16*)carve(SB * 1024 * 2);          // layer input (f16)
  _Float16* gx     = (_Float16*)carve(2 * SB * 2048 * 2);      // [dir][S*B][4H]
  _Float16* hcat   = (_Float16*)carve(SB * 1024 * 2);          // [S*B][2H]
  _Float16* hstate = (_Float16*)carve(2 * 2 * (size_t)B * 512 * 2);
  unsigned int* cnt = (unsigned int*)carve(256);
  _Float16* Wt0f = (_Float16*)carve((size_t)512 * 2048 * 2);
  _Float16* Wt0b = (_Float16*)carve((size_t)512 * 2048 * 2);
  _Float16* Wt1f = (_Float16*)carve((size_t)512 * 2048 * 2);
  _Float16* Wt1b = (_Float16*)carve((size_t)512 * 2048 * 2);
  _Float16* Ut0f = (_Float16*)carve((size_t)512 * 2048 * 2);
  _Float16* Ut0b = (_Float16*)carve((size_t)512 * 2048 * 2);
  _Float16* Ut1f = (_Float16*)carve((size_t)1024 * 2048 * 2);
  _Float16* Ut1b = (_Float16*)carve((size_t)1024 * 2048 * 2);
  _Float16* Wyt0 = (_Float16*)carve((size_t)1024 * 1024 * 2);
  _Float16* Wyt1 = (_Float16*)carve((size_t)1024 * 1024 * 2);

  auto T = [&](const float* in, _Float16* out, int K, int N) {
    int total = K * N;
    transpose_f32_to_f16_kernel<<<(total + 255) / 256, 256, 0, stream>>>(in, out, K, N);
  };
  T(W0f, Wt0f, 512, 2048);  T(U0f, Ut0f, 512, 2048);
  T(W0b, Wt0b, 512, 2048);  T(U0b, Ut0b, 512, 2048);
  T(Wy0, Wyt0, 1024, 1024);
  T(W1f, Wt1f, 512, 2048);  T(U1f, Ut1f, 1024, 2048);
  T(W1b, Wt1b, 512, 2048);  T(U1b, Ut1b, 1024, 2048);
  T(Wy1, Wyt1, 1024, 1024);

  // embedding gather -> xs [S,B,E] f16
  {
    size_t total = SB * E;
    embed_kernel<<<(unsigned)((total + 255) / 256), 256, 0, stream>>>(src, embed, xs, S, B, E);
  }

  auto GEMM = [&](const _Float16* A, const _Float16* Bt, const float* bias, void* out,
                  int N, int K, int lda, int ldo, int mode) {
    dim3 grid(N / 128, M / 64);
    gemm_bias_kernel<<<grid, 256, 0, stream>>>(A, Bt, bias, out, M, N, K, lda, ldo, mode, S, B);
  };

  const size_t recur_lds = (size_t)128 * WLD * sizeof(_Float16);  // 130KB dynamic LDS

  // ---------------- layer 0 ----------------
  GEMM(xs, Ut0f, b0f, gx,                 2048, 512, 512, 2048, 0);
  GEMM(xs, Ut0b, b0b, gx + SB * 2048,     2048, 512, 512, 2048, 0);
  init_state_kernel<<<(131072 + 255) / 256, 256, 0, stream>>>(hstate, cnt, 2 * 2 * B * 512);
  lstm_recur_kernel<<<dim3(GPD, 2), 256, recur_lds, stream>>>(gx, Wt0f, Wt0b, hstate, hcat, cnt, S, B);
  GEMM(hcat, Wyt0, by0, xs,               1024, 1024, 1024, 1024, 0);

  // ---------------- layer 1 ----------------
  GEMM(xs, Ut1f, b1f, gx,                 2048, 1024, 1024, 2048, 0);
  GEMM(xs, Ut1b, b1b, gx + SB * 2048,     2048, 1024, 1024, 2048, 0);
  init_state_kernel<<<(131072 + 255) / 256, 256, 0, stream>>>(hstate, cnt, 2 * 2 * B * 512);
  lstm_recur_kernel<<<dim3(GPD, 2), 256, recur_lds, stream>>>(gx, Wt1f, Wt1b, hstate, hcat, cnt, S, B);

  // final projection with fused [S,B,.] -> [B,S,.] transpose, f32 to d_out
  GEMM(hcat, Wyt1, by1, d_out,            1024, 1024, 1024, 1024, 1);

  // second output: last hcat -> d_out tail
  float* out2 = (float*)d_out + SB * 1024;
  last_hcat_kernel<<<(B * 1024 + 255) / 256, 256, 0, stream>>>(hcat, out2, S, B);
}
